// BayesVGGLike__32315333935353
// MI455X (gfx1250) — matrix operand order repaired and checked by reading, not compiled, not run on hardware
//
#include <hip/hip_runtime.h>
#include <hip/hip_bf16.h>
#include <math.h>

// ---------------------------------------------------------------------------
// Types for CDNA5 WMMA (wave32): 16x16x32 bf16 -> f32 accum
// ---------------------------------------------------------------------------
typedef __attribute__((ext_vector_type(16))) __bf16        bf16x16;
typedef __attribute__((ext_vector_type(8)))  float         f32x8;
typedef __attribute__((ext_vector_type(4)))  float         f32x4;
typedef __attribute__((ext_vector_type(8)))  unsigned int  u32x8;
typedef __attribute__((ext_vector_type(4)))  unsigned int  u32x4;
typedef __attribute__((address_space(3)))    unsigned short lds_ushort;

static __device__ __forceinline__ unsigned short f2bf(float f) {
    unsigned int u = __float_as_uint(f);
    unsigned int r = u + 0x7FFFu + ((u >> 16) & 1u);     // round-to-nearest-even
    if ((u & 0x7F800000u) == 0x7F800000u) r = u;          // keep Inf/NaN
    return (unsigned short)(r >> 16);
}
// hot path (activations are finite): RNE without NaN guard
static __device__ __forceinline__ unsigned int f2bf_fast(float f) {
    unsigned int u = __float_as_uint(f);
    return (u + 0x7FFFu + ((u >> 16) & 1u)) >> 16;
}

// ---------------------------------------------------------------------------
// Weight conversion into PADDED [Mpad x Kpad] bf16 buffers (pad = 0):
//   wm -> bf16(w), wm2 -> bf16(w^2), wv -> bf16(1 - w^2)
// ---------------------------------------------------------------------------
__global__ void k_cvtw(const float* __restrict__ w,
                       unsigned short* __restrict__ wm,
                       unsigned short* __restrict__ wm2,
                       unsigned short* __restrict__ wv,
                       int M, int K, int Kpad, int total) {
    int i = blockIdx.x * blockDim.x + threadIdx.x;
    if (i >= total) return;
    int m = i / Kpad, k = i - m * Kpad;
    bool in = (m < M) && (k < K);
    float x = in ? w[(size_t)m * K + k] : 0.0f;
    wm[i]  = f2bf(x);
    wm2[i] = f2bf(x * x);
    wv[i]  = in ? f2bf(1.0f - x * x) : (unsigned short)0;
}

// x2 = m*m (+ v)
__global__ void k_sq(const float* __restrict__ m, const float* __restrict__ v,
                     float* __restrict__ x2, long long n, int hasV) {
    long long i = (long long)blockIdx.x * blockDim.x + threadIdx.x;
    if (i >= n) return;
    float mm = m[i];
    x2[i] = mm * mm + (hasV ? v[i] : 0.0f);
}

__global__ void k_zero(float* p, int n) {
    int i = blockIdx.x * blockDim.x + threadIdx.x;
    if (i < n) p[i] = 0.0f;
}

// ---------------------------------------------------------------------------
// WMMA GEMM:  O[M x N] (+)= A_bf16[Mpad x Kpad] * B(k,n)
//   mode 0: B(k,n) = implicit im2col of NCHW f32 tensor X (3x3, pad 1)
//   mode 1: B(k,n) = X[n*K + k]   (dense, X row-major [N x K], K mult of 64)
// Block tile 128x64, 8 waves (4x2), each wave 2x2 tiles of 16x16, K-step 64.
// A tile is staged with the CDNA5 async global->LDS engine (ASYNCcnt);
// B tile is gathered/converted through VGPRs (im2col + f32->bf16).
// ---------------------------------------------------------------------------
#define BM 128
#define BN 64
#define BK 64
#define LDSK 72   // padded k-stride (ushort units): 144B rows, 16B-aligned

__global__ __launch_bounds__(256)
void k_gemm(const unsigned short* __restrict__ A,
            const float* __restrict__ X,
            float* __restrict__ O,
            int M, int N, int K, int Kpad,
            int mode, int accum,
            int Cin, int H, int W) {
    __shared__ __align__(16) unsigned short sA[BM * LDSK];
    __shared__ __align__(16) unsigned short sB[BN * LDSK];

    const int tid  = threadIdx.x;
    const int lane = tid & 31;
    const int wave = tid >> 5;
    const int wm0  = (wave & 3) << 5;   // 4 waves along M (32 rows each)
    const int wn0  = (wave >> 2) << 5;  // 2 waves along N (32 cols each)
    const int l16  = lane & 15;
    const int lh   = lane >> 4;

    const int bn0 = blockIdx.x * BN;
    const int bm0 = blockIdx.y * BM;
    const int HW  = H * W;

    // LDS byte-offset base of sA (raw AS(3) offset for async-to-LDS dst)
    const unsigned sA_base = (unsigned)(unsigned long long)(lds_ushort*)&sA[0];

    // ---- per-thread B-column decode (invariant across the K loop) ----
    const int nloc = tid & 63;            // column within tile
    const int kb   = (tid >> 6) << 4;     // 16 consecutive k per thread
    const int gn   = bn0 + nloc;
    int bb = 0, yy = 0, xx = 0;
    if (mode == 0) {
        bb = gn / HW; int s = gn - bb * HW;
        yy = s / W;   xx = s - yy * W;
    }

    f32x8 acc[2][2] = {};

    const int KT = Kpad / BK;
    for (int kt = 0; kt < KT; ++kt) {
        const int k0 = kt * BK;

        // ---- stage A tile (128 x 64 bf16): async global->LDS, 4x b128/thread
        #pragma unroll
        for (int i = 0; i < 4; ++i) {
            int e  = i * 256 + tid;       // b128 chunk index within tile
            int m  = e >> 3;              // 8 chunks per 64-elem row
            int kq = (e & 7) << 3;        // elem offset
            unsigned long long gp = (unsigned long long)(size_t)
                &A[(size_t)(bm0 + m) * Kpad + k0 + kq];
            unsigned lds = sA_base + (unsigned)(m * LDSK + kq) * 2u;
            asm volatile("global_load_async_to_lds_b128 %0, %1, off"
                         :: "v"(lds), "v"(gp) : "memory");
        }
        if (kt + 1 < KT)   // hint next A tile toward the caches
            __builtin_prefetch(&A[(size_t)bm0 * Kpad + (k0 + BK)], 0, 1);

        // ---- stage B tile (64k x 64n), stored [n][k] ----
        {
            unsigned int pk[8];
            const int gk0 = k0 + kb;
            if (mode == 1) {
                const float* src = X + (size_t)gn * K + gk0;
                #pragma unroll
                for (int i = 0; i < 4; ++i) {
                    f32x4 fv = *(const f32x4*)&src[i * 4];
                    pk[2 * i]     = f2bf_fast(fv[0]) | (f2bf_fast(fv[1]) << 16);
                    pk[2 * i + 1] = f2bf_fast(fv[2]) | (f2bf_fast(fv[3]) << 16);
                }
            } else {
                int ci = gk0 / 9, rs = gk0 - ci * 9;
                unsigned short hs[16];
                #pragma unroll
                for (int i = 0; i < 16; ++i) {
                    int r = rs / 3, q = rs - r * 3;
                    int iy = yy + r - 1, ix = xx + q - 1;
                    bool ok = (gk0 + i < K) & (iy >= 0) & (iy < H) &
                              (ix >= 0) & (ix < W);
                    size_t idx = ok ? ((((size_t)bb * Cin + ci) * H + iy) * W + ix)
                                    : (size_t)0;
                    float f = X[idx];             // clamped address, always valid
                    f = ok ? f : 0.0f;
                    hs[i] = (unsigned short)f2bf_fast(f);
                    if (++rs == 9) { rs = 0; ++ci; }
                }
                #pragma unroll
                for (int i = 0; i < 8; ++i)
                    pk[i] = (unsigned int)hs[2 * i] |
                            ((unsigned int)hs[2 * i + 1] << 16);
            }
            u32x4* dst = (u32x4*)&sB[nloc * LDSK + kb];
            dst[0] = u32x4{pk[0], pk[1], pk[2], pk[3]};
            dst[1] = u32x4{pk[4], pk[5], pk[6], pk[7]};
        }
        // async A-tile transfers must be complete before fragment reads
        asm volatile("s_wait_asynccnt 0" ::: "memory");
        __syncthreads();

        // ---- two 32-wide WMMA sub-steps per staged 64-wide K tile ----
        #pragma unroll
        for (int sub = 0; sub < 2; ++sub) {
            const int ksub = sub << 5;
            bf16x16 afrag[2], bfrag[2];
            #pragma unroll
            for (int mi = 0; mi < 2; ++mi) {
                int row = wm0 + mi * 16 + l16;
                union { u32x8 u; bf16x16 h; } c;
                #pragma unroll
                for (int v = 0; v < 8; ++v) {
                    int k = ksub + ((v & 3) << 1) + (lh << 3) + ((v >> 2) << 4);
                    c.u[v] = *(const unsigned int*)&sA[row * LDSK + k];
                }
                afrag[mi] = c.h;
            }
            #pragma unroll
            for (int ni = 0; ni < 2; ++ni) {
                int col = wn0 + ni * 16 + l16;
                union { u32x8 u; bf16x16 h; } c;
                #pragma unroll
                for (int v = 0; v < 8; ++v) {
                    int k = ksub + ((v & 3) << 1) + (lh << 3) + ((v >> 2) << 4);
                    c.u[v] = *(const unsigned int*)&sB[col * LDSK + k];
                }
                bfrag[ni] = c.h;
            }
            #pragma unroll
            for (int mi = 0; mi < 2; ++mi)
                #pragma unroll
                for (int ni = 0; ni < 2; ++ni)
                    acc[mi][ni] = __builtin_amdgcn_wmma_f32_16x16x32_bf16(
                        false, afrag[mi], false, bfrag[ni],
                        (short)0, acc[mi][ni], false, false);
        }
        __syncthreads();
    }

    // ---- epilogue (C/D layout: l16 = N col, lh selects M+8, vgpr r = M) ----
    #pragma unroll
    for (int mi = 0; mi < 2; ++mi)
        #pragma unroll
        for (int ni = 0; ni < 2; ++ni)
            #pragma unroll
            for (int r = 0; r < 8; ++r) {
                int row = bm0 + wm0 + mi * 16 + r + (lh << 3);
                int col = bn0 + wn0 + ni * 16 + l16;
                if (row < M) {
                    size_t idx;
                    if (mode == 1) {
                        idx = (size_t)col * M + row;
                    } else {
                        int b = col / HW, s = col - b * HW;
                        idx = ((size_t)b * M + row) * HW + s;
                    }
                    float val = acc[mi][ni][r];
                    if (accum) val += O[idx];
                    O[idx] = val;
                }
            }
}

// ---------------------------------------------------------------------------
// Per-channel stats: sum(m), sum(m^2), sum(v) over (B, HW)
// ---------------------------------------------------------------------------
__global__ __launch_bounds__(256)
void k_stats(const float* __restrict__ om, const float* __restrict__ ov,
             float* __restrict__ stats, int C, int HW, int total) {
    int c = blockIdx.x;
    float sm = 0.f, sm2 = 0.f, sv = 0.f;
    for (int t = blockIdx.y * blockDim.x + threadIdx.x; t < total;
         t += gridDim.y * blockDim.x) {
        int b = t / HW, s = t - b * HW;
        size_t idx = ((size_t)b * C + c) * HW + s;
        float m = om[idx], v = ov[idx];
        sm += m; sm2 += m * m; sv += v;
    }
    __shared__ float red[3][256];
    red[0][threadIdx.x] = sm; red[1][threadIdx.x] = sm2; red[2][threadIdx.x] = sv;
    __syncthreads();
    for (int off = 128; off > 0; off >>= 1) {
        if ((int)threadIdx.x < off) {
            red[0][threadIdx.x] += red[0][threadIdx.x + off];
            red[1][threadIdx.x] += red[1][threadIdx.x + off];
            red[2][threadIdx.x] += red[2][threadIdx.x + off];
        }
        __syncthreads();
    }
    if (threadIdx.x == 0) {
        atomicAdd(&stats[c],         red[0][0]);
        atomicAdd(&stats[C + c],     red[1][0]);
        atomicAdd(&stats[2 * C + c], red[2][0]);
    }
}

// Fused bias(bv) + BN + erf-sign. (bm cancels in mean-subtraction; bv=1-bm^2
// enters the variance and v.)  In-place on (m, v).
__global__ void k_bnsign(float* __restrict__ m, float* __restrict__ v,
                         const float* __restrict__ stats,
                         const float* __restrict__ g, const float* __restrict__ be,
                         const float* __restrict__ bm,
                         int C, int HW, long long count, float cnt) {
    long long i = (long long)blockIdx.x * blockDim.x + threadIdx.x;
    if (i >= count) return;
    int c = (int)((i / HW) % C);
    float mu   = stats[c] / cnt;
    float em2  = stats[C + c] / cnt;
    float bmv  = bm[c];
    float bv   = 1.0f - bmv * bmv;
    float mnv  = stats[2 * C + c] / cnt + bv;
    float var  = em2 - mu * mu + mnv;
    float inv  = g[c] * rsqrtf(var + 1e-5f);
    float mm   = (m[i] - mu) * inv + be[c];
    float vv   = (v[i] + bv) * inv * inv;
    float mn   = erff(mm * rsqrtf(2.0f * (vv + 1e-8f)));
    m[i] = mn;
    v[i] = 1.0f - mn * mn;
}

// 2x2 mean pool: m -> mean, v -> mean/4
__global__ void k_pool(const float* __restrict__ mi_, const float* __restrict__ vi_,
                       float* __restrict__ mo, float* __restrict__ vo,
                       int Bn, int C, int H, int W) {
    int Ho = H >> 1, Wo = W >> 1;
    long long total = (long long)Bn * C * Ho * Wo;
    long long i = (long long)blockIdx.x * blockDim.x + threadIdx.x;
    if (i >= total) return;
    int wo = (int)(i % Wo); long long t = i / Wo;
    int ho = (int)(t % Ho); t /= Ho;
    int c  = (int)(t % C);  int b = (int)(t / C);
    size_t base = (((size_t)b * C + c) * H + ho * 2) * W + wo * 2;
    mo[i] = 0.25f   * (mi_[base] + mi_[base + 1] + mi_[base + W] + mi_[base + W + 1]);
    vo[i] = 0.0625f * (vi_[base] + vi_[base + 1] + vi_[base + W] + vi_[base + W + 1]);
}

// Final: add fc2 bias terms, probit adjust, log-softmax, prob + NLL
__global__ __launch_bounds__(256)
void k_final(const float* __restrict__ om, const float* __restrict__ ov,
             const float* __restrict__ bf2, const int* __restrict__ tgt,
             float* __restrict__ out) {
    __shared__ float red[256];
    int b = threadIdx.x;                       // 256 batch rows
    float adj[10];
    float mx = -1e30f;
    for (int c = 0; c < 10; ++c) {
        float bbv = bf2[c];
        float m = om[b * 10 + c] + bbv;
        float v = ov[b * 10 + c] + 1.0f - bbv * bbv;
        float z  = m * (1.0f / 32.0f);         // / sqrt(1024)
        float zv = v * (1.0f / 1024.0f);
        float a  = z * rsqrtf(1.0f + 0.39269908169872414f * zv);  // pi/8
        adj[c] = a;
        mx = fmaxf(mx, a);
    }
    float se = 0.f;
    for (int c = 0; c < 10; ++c) se += expf(adj[c] - mx);
    float lse = mx + logf(se);
    int t = tgt[b];
    float nll = 0.f;
    for (int c = 0; c < 10; ++c) {
        float lp = adj[c] - lse;
        out[b * 10 + c] = expf(lp);
        if (c == t) nll -= lp;
    }
    red[b] = nll;
    __syncthreads();
    for (int off = 128; off > 0; off >>= 1) {
        if (b < off) red[b] += red[b + off];
        __syncthreads();
    }
    if (b == 0) out[2560] = red[0];
}

// ---------------------------------------------------------------------------
// Host orchestration
// ---------------------------------------------------------------------------
static inline int statsGridY(int total) {
    int g = (total + 2047) / 2048;
    if (g < 1) g = 1; if (g > 64) g = 64;
    return g;
}

static void gemm(const unsigned short* A, const float* X, float* O,
                 int M, int N, int K, int Kpad, int mode, int accum,
                 int Cin, int H, int W, hipStream_t s) {
    dim3 grid((N + BN - 1) / BN, (M + BM - 1) / BM);
    k_gemm<<<grid, 256, 0, s>>>(A, X, O, M, N, K, Kpad, mode, accum, Cin, H, W);
}

static void bn_sign(float* m, float* v, float* stats,
                    const float* g, const float* be, const float* bm,
                    int Bn, int C, int HW, hipStream_t s) {
    int total = Bn * HW;
    k_zero<<<(3 * C + 255) / 256, 256, 0, s>>>(stats, 3 * C);
    dim3 sg(C, statsGridY(total));
    k_stats<<<sg, 256, 0, s>>>(m, v, stats, C, HW, total);
    long long count = (long long)Bn * C * HW;
    k_bnsign<<<(unsigned)((count + 255) / 256), 256, 0, s>>>(
        m, v, stats, g, be, bm, C, HW, count, (float)total);
}

extern "C" void kernel_launch(void* const* d_in, const int* in_sizes, int n_in,
                              void* d_out, int out_size, void* d_ws, size_t ws_size,
                              hipStream_t stream) {
    (void)in_sizes; (void)n_in; (void)out_size; (void)ws_size;
    const int Bn = 256;
    const float* inputs  = (const float*)d_in[0];
    const int*   targets = (const int*)d_in[1];
    const float* w[8];  const float* bb[8];  const float* gg[6];  const float* bee[6];
    for (int l = 0; l < 6; ++l) {
        w[l]   = (const float*)d_in[2 + 4 * l + 0];
        bb[l]  = (const float*)d_in[2 + 4 * l + 1];
        gg[l]  = (const float*)d_in[2 + 4 * l + 2];
        bee[l] = (const float*)d_in[2 + 4 * l + 3];
    }
    const float* wf1 = (const float*)d_in[26]; const float* bf1 = (const float*)d_in[27];
    const float* g41 = (const float*)d_in[28]; const float* be41 = (const float*)d_in[29];
    const float* wf2 = (const float*)d_in[30]; const float* bf2 = (const float*)d_in[31];
    w[6] = wf1; bb[6] = bf1; w[7] = wf2; bb[7] = bf2;

    // ---- workspace carve ----
    const size_t ACT = 33554432ull;      // floats: 256*128*32*32
    float* fAm = (float*)d_ws;
    float* fAv = fAm + ACT;
    float* fBm = fAv + ACT;
    float* fBv = fBm + ACT;
    float* fX2 = fBv + ACT;
    float* stats = fX2 + ACT;            // 3*1024 max
    unsigned short* wb = (unsigned short*)(stats + 4096);

    // bf16 weight triples, zero-padded to [Mpad x Kpad]
    static const int wM[8]  = {128, 128, 256, 256, 512, 512, 1024, 10};
    static const int wK[8]  = {27, 1152, 1152, 2304, 2304, 4608, 8192, 1024};
    static const int wMp[8] = {128, 128, 256, 256, 512, 512, 1024, 128};
    static const int wKp[8] = {64, 1152, 1152, 2304, 2304, 4608, 8192, 1024};
    const unsigned short *wm[8], *wm2[8], *wv[8];
    size_t off = 0;
    for (int l = 0; l < 8; ++l) {
        int total = wMp[l] * wKp[l];
        unsigned short* base = wb + off;
        wm[l] = base; wm2[l] = base + total; wv[l] = base + 2 * (size_t)total;
        off += 3 * (size_t)total;
        k_cvtw<<<(total + 255) / 256, 256, 0, stream>>>(
            w[l], (unsigned short*)wm[l], (unsigned short*)wm2[l],
            (unsigned short*)wv[l], wM[l], wK[l], wKp[l], total);
    }

    struct LCfg { int Cin, H, W, Cout, pool; };
    const LCfg cfg[6] = {
        {3,   32, 32, 128, 0}, {128, 32, 32, 128, 1},
        {128, 16, 16, 256, 0}, {256, 16, 16, 256, 1},
        {256,  8,  8, 512, 0}, {512,  8,  8, 512, 1},
    };

    float *curM = fBm, *curV = fBv, *nxtM = fAm, *nxtV = fAv;

    for (int l = 0; l < 6; ++l) {
        const LCfg& c = cfg[l];
        int HW = c.H * c.W, N = Bn * HW, K = c.Cin * 9, Kp = wKp[l], M = c.Cout;
        const float* mIn = (l == 0) ? inputs : curM;
        const float* vIn = (l == 0) ? nullptr : curV;
        long long inCount = (long long)Bn * c.Cin * HW;
        k_sq<<<(unsigned)((inCount + 255) / 256), 256, 0, stream>>>(
            mIn, vIn, fX2, inCount, l == 0 ? 0 : 1);
        gemm(wm[l], mIn, nxtM, M, N, K, Kp, 0, 0, c.Cin, c.H, c.W, stream);
        if (l == 0) {
            gemm(wv[l], fX2, nxtV, M, N, K, Kp, 0, 0, c.Cin, c.H, c.W, stream);
        } else {
            gemm(wm2[l], vIn, nxtV, M, N, K, Kp, 0, 0, c.Cin, c.H, c.W, stream);
            gemm(wv[l],  fX2, nxtV, M, N, K, Kp, 0, 1, c.Cin, c.H, c.W, stream);
        }
        bn_sign(nxtM, nxtV, stats, gg[l], bee[l], bb[l], Bn, M, HW, stream);
        if (c.pool) {
            long long outCount = (long long)Bn * M * (HW / 4);
            k_pool<<<(unsigned)((outCount + 255) / 256), 256, 0, stream>>>(
                nxtM, nxtV, curM, curV, Bn, M, c.H, c.W);
        } else {
            float* t;
            t = curM; curM = nxtM; nxtM = t;
            t = curV; curV = nxtV; nxtV = t;
        }
    }
    // cur* = flattened [256 x 8192]

    // ---- FC1: M=1024, K=8192, N=256 (dense) ----
    {
        long long n = (long long)Bn * 8192;
        k_sq<<<(unsigned)((n + 255) / 256), 256, 0, stream>>>(curM, curV, fX2, n, 1);
        gemm(wm[6],  curM, nxtM, 1024, Bn, 8192, 8192, 1, 0, 0, 1, 1, stream);
        gemm(wm2[6], curV, nxtV, 1024, Bn, 8192, 8192, 1, 0, 0, 1, 1, stream);
        gemm(wv[6],  fX2,  nxtV, 1024, Bn, 8192, 8192, 1, 1, 0, 1, 1, stream);
        bn_sign(nxtM, nxtV, stats, g41, be41, bf1, Bn, 1024, 1, stream);
    }
    // ---- FC2: M=10, K=1024, N=256 (dense) ----
    {
        long long n = (long long)Bn * 1024;
        k_sq<<<(unsigned)((n + 255) / 256), 256, 0, stream>>>(nxtM, nxtV, fX2, n, 1);
        gemm(wm[7],  nxtM, curM, 10, Bn, 1024, 1024, 1, 0, 0, 1, 1, stream);
        gemm(wm2[7], nxtV, curV, 10, Bn, 1024, 1024, 1, 0, 0, 1, 1, stream);
        gemm(wv[7],  fX2,  curV, 10, Bn, 1024, 1024, 1, 1, 0, 1, 1, stream);
    }
    k_final<<<1, 256, 0, stream>>>(curM, curV, bf2, targets, (float*)d_out);
}
